// Alignment_46669114638742
// MI455X (gfx1250) — matrix operand order repaired and checked by reading, not compiled, not run on hardware
//
#include <hip/hip_runtime.h>
#include <hip/hip_bf16.h>

// ---------------------------------------------------------------------------
// Bahdanau additive attention, fused for MI455X (gfx1250, wave32, WMMA bf16)
// B=16, S=4096, E=1024, H=512
// ---------------------------------------------------------------------------

#define B_   16
#define S_   4096
#define E_   1024
#define H_   512

typedef __attribute__((ext_vector_type(16))) __bf16 v16bf;
typedef __attribute__((ext_vector_type(8)))  __bf16 v8bf;
typedef __attribute__((ext_vector_type(8)))  float  v8f;

__device__ __forceinline__ unsigned short f2bf(float f) {
    union { float f; unsigned u; } v; v.f = f;
    unsigned r = v.u + 0x7FFFu + ((v.u >> 16) & 1u);   // round-to-nearest-even
    return (unsigned short)(r >> 16);
}

// ---------------------------------------------------------------------------
// Kernel 0: repack We [1024 x 512] f32 -> bf16 B-fragment layout.
// Web[kt][nt][lane][16]: lane 0-15 holds col n=nt*16+lane, k = kt*32 + 0..15
//                        lane 16-31 holds col n, k = kt*32 + 16..31
// ---------------------------------------------------------------------------
__global__ void webprep_kernel(const float* __restrict__ We,
                               unsigned short* __restrict__ web) {
    const int kt   = blockIdx.x >> 5;
    const int nt   = blockIdx.x & 31;
    const int lane = threadIdx.x;          // 32 threads = 1 wave
    const int kb   = kt * 32 + (lane >> 4) * 16;
    const int n    = nt * 16 + (lane & 15);
    unsigned short* dst = web + (((kt * 32 + nt) * 32) + lane) * 16;
#pragma unroll
    for (int j = 0; j < 16; ++j)
        dst[j] = f2bf(We[(kb + j) * H_ + n]);
}

// ---------------------------------------------------------------------------
// Kernel 1: dec_proj[b,h] = decoder_hidden[b,:] @ Wd + bd   (16 x 512)
// ---------------------------------------------------------------------------
__global__ __launch_bounds__(256) void decproj_kernel(
        const float* __restrict__ dh, const float* __restrict__ Wd,
        const float* __restrict__ bd, float* __restrict__ decp) {
    const int b = blockIdx.x, tid = threadIdx.x;
    const float* dhb = dh + b * E_;
    float a0 = 0.f, a1 = 0.f;
    for (int e = 0; e < E_; ++e) {
        float d = dhb[e];                     // wave-uniform -> scalar load
        a0 += d * Wd[e * H_ + tid];
        a1 += d * Wd[e * H_ + tid + 256];
    }
    decp[b * H_ + tid]       = a0 + bd[tid];
    decp[b * H_ + tid + 256] = a1 + bd[tid + 256];
}

// ---------------------------------------------------------------------------
// Kernel 2: fused  scores[b,s] = tanh(enc@We + be + decp) . Wo + bo
// 1024 blocks x 256 threads (8 waves). Block owns 64 rows; wave owns 64 cols.
// 16 WMMAs (4 M-tiles x 4 N-tiles) per K-step, K = 32 steps of 32.
// ---------------------------------------------------------------------------
#define TM  64        // rows per block
#define AST 40        // LDS row stride (bf16 elems) -> bank-conflict free

__global__ __launch_bounds__(256) void scores_kernel(
        const float* __restrict__ enc,            // [B*S, 1024]
        const unsigned short* __restrict__ web,   // bf16 fragment layout
        const float* __restrict__ be,
        const float* __restrict__ decp,           // [B, 512]
        const float* __restrict__ Wo,             // [512]
        const float* __restrict__ bo,
        float* __restrict__ scores) {             // [B*S]
    __shared__ unsigned short As[TM * AST];       // 64x32 bf16 chunk (padded)
    __shared__ float swl[8 * TM];                 // per-wave partial scores

    const int tid  = threadIdx.x;
    const int lane = tid & 31;
    const int w    = tid >> 5;                    // wave id 0..7
    const long row0 = (long)blockIdx.x * TM;      // 64 rows, same batch b
    const int  b    = (int)(row0 >> 12);          // row0 / 4096

    v8f acc[4][4];
#pragma unroll
    for (int mt = 0; mt < 4; ++mt)
#pragma unroll
        for (int t = 0; t < 4; ++t) acc[mt][t] = (v8f){};

    // staging coords: 4 threads per row, 2x float4 each -> 32 f32 per row
    const int sr = tid >> 2;                      // 0..63
    const int c8 = (tid & 3) << 3;                // 0,8,16,24
    // A-fragment coords (contiguous k order: k = 16*(lane/16) + s, matches web)
    const int k0 = (lane >> 4) * 16;
    const int rA = lane & 15;

    for (int kt = 0; kt < 32; ++kt) {
        // ---- stage 64x32 f32 -> bf16 LDS ----
        const float* src = enc + (row0 + sr) * E_ + kt * 32 + c8;
        const float4 v0 = *(const float4*)(src);
        const float4 v1 = *(const float4*)(src + 4);
        uint4 pk;
        pk.x = (unsigned)f2bf(v0.x) | ((unsigned)f2bf(v0.y) << 16);
        pk.y = (unsigned)f2bf(v0.z) | ((unsigned)f2bf(v0.w) << 16);
        pk.z = (unsigned)f2bf(v1.x) | ((unsigned)f2bf(v1.y) << 16);
        pk.w = (unsigned)f2bf(v1.z) | ((unsigned)f2bf(v1.w) << 16);
        *(uint4*)(&As[sr * AST + c8]) = pk;
        __syncthreads();

        // ---- build A fragments for the four 16-row tiles ----
        v16bf a[4];
#pragma unroll
        for (int mt = 0; mt < 4; ++mt) {
            const unsigned short* ap = &As[(mt * 16 + rA) * AST + k0];
            v8bf lo = *reinterpret_cast<const v8bf*>(ap);        // K k0..k0+7
            v8bf hi = *reinterpret_cast<const v8bf*>(ap + 8);    // K k0+8..+15
#pragma unroll
            for (int j = 0; j < 8; ++j) { a[mt][j] = lo[j]; a[mt][j + 8] = hi[j]; }
        }

        // ---- 4 N-tiles x 4 M-tiles = 16 WMMAs ----
#pragma unroll
        for (int t = 0; t < 4; ++t) {
            const int nt = w * 4 + t;
            const v16bf bf = *reinterpret_cast<const v16bf*>(
                web + ((kt * 32 + nt) * 32 + lane) * 16);
#pragma unroll
            for (int mt = 0; mt < 4; ++mt)
                acc[mt][t] = __builtin_amdgcn_wmma_f32_16x16x32_bf16(
                    false, a[mt], false, bf, (short)0, acc[mt][t], false, false);
        }
        __syncthreads();
    }

    // ---- epilogue: tanh(+be+decp) . Wo, reduce over H ----
    const float* dp = decp + b * H_;
    float p[4][8];
#pragma unroll
    for (int mt = 0; mt < 4; ++mt)
#pragma unroll
        for (int i = 0; i < 8; ++i) p[mt][i] = 0.f;

#pragma unroll
    for (int t = 0; t < 4; ++t) {
        const int n  = w * 64 + t * 16 + (lane & 15);
        const float add = be[n] + dp[n];
        const float won = Wo[n];
#pragma unroll
        for (int mt = 0; mt < 4; ++mt)
#pragma unroll
            for (int i = 0; i < 8; ++i)
                p[mt][i] += tanhf(acc[mt][t][i] + add) * won;
    }
    // butterfly over the 16-lane half (C/D tile: lane%16 = N, lane/16 picks M+8)
#pragma unroll
    for (int mt = 0; mt < 4; ++mt)
#pragma unroll
        for (int i = 0; i < 8; ++i) {
            float v = p[mt][i];
            for (int off = 1; off < 16; off <<= 1) v += __shfl_xor(v, off);
            p[mt][i] = v;
        }
    if ((lane & 15) == 0) {
        const int half = lane >> 4;
#pragma unroll
        for (int mt = 0; mt < 4; ++mt)
#pragma unroll
            for (int i = 0; i < 8; ++i)
                swl[w * TM + mt * 16 + half * 8 + i] = p[mt][i];
    }
    __syncthreads();
    if (tid < TM) {
        float s = bo[0];
#pragma unroll
        for (int ww = 0; ww < 8; ++ww) s += swl[ww * TM + tid];
        scores[row0 + tid] = s;
    }
}

// ---------------------------------------------------------------------------
// Kernel 3: softmax over S=4096 per batch row
// ---------------------------------------------------------------------------
__global__ __launch_bounds__(256) void softmax_kernel(
        const float* __restrict__ scores, float* __restrict__ attn) {
    __shared__ float red[256];
    const int b = blockIdx.x, tid = threadIdx.x;
    const float* sp = scores + b * S_;
    float x[16], mx = -3.0e38f;
#pragma unroll
    for (int j = 0; j < 16; ++j) { x[j] = sp[tid + j * 256]; mx = fmaxf(mx, x[j]); }
    red[tid] = mx; __syncthreads();
    for (int s = 128; s > 0; s >>= 1) {
        if (tid < s) red[tid] = fmaxf(red[tid], red[tid + s]);
        __syncthreads();
    }
    mx = red[0]; __syncthreads();
    float sum = 0.f;
#pragma unroll
    for (int j = 0; j < 16; ++j) { x[j] = expf(x[j] - mx); sum += x[j]; }
    red[tid] = sum; __syncthreads();
    for (int s = 128; s > 0; s >>= 1) {
        if (tid < s) red[tid] += red[tid + s];
        __syncthreads();
    }
    const float inv = 1.0f / red[0];
#pragma unroll
    for (int j = 0; j < 16; ++j) attn[b * S_ + tid + j * 256] = x[j] * inv;
}

// ---------------------------------------------------------------------------
// Kernel 4+5: context[b,e] = sum_s attn[b,s]*enc[b,s,e]  (deterministic 2-stage)
// ---------------------------------------------------------------------------
__global__ __launch_bounds__(256) void ctx_partial_kernel(
        const float* __restrict__ enc, const float* __restrict__ attn,
        float* __restrict__ part) {
    const int ec = blockIdx.x, sc = blockIdx.y, b = blockIdx.z;
    const int e = ec * 256 + threadIdx.x;
    const long base = (long)b * S_ + sc * 256;
    float acc = 0.f;
    for (int s = 0; s < 256; ++s) {
        const float a = attn[b * S_ + sc * 256 + s];   // uniform -> scalar load
        acc += a * enc[(base + s) * E_ + e];
    }
    part[((long)sc * B_ + b) * E_ + e] = acc;
}

__global__ __launch_bounds__(256) void ctx_reduce_kernel(
        const float* __restrict__ part, float* __restrict__ out) {
    const int idx = blockIdx.x * 256 + threadIdx.x;    // 16384 = B*E
    float s = 0.f;
#pragma unroll
    for (int sc = 0; sc < 16; ++sc) s += part[sc * (B_ * E_) + idx];
    out[idx] = s;
}

// ---------------------------------------------------------------------------
extern "C" void kernel_launch(void* const* d_in, const int* in_sizes, int n_in,
                              void* d_out, int out_size, void* d_ws, size_t ws_size,
                              hipStream_t stream) {
    const float* enc = (const float*)d_in[0];   // [16,4096,1024]
    const float* dh  = (const float*)d_in[1];   // [16,1024]
    const float* We  = (const float*)d_in[2];   // [1024,512]
    const float* be  = (const float*)d_in[3];   // [512]
    const float* Wd  = (const float*)d_in[4];   // [1024,512]
    const float* bd  = (const float*)d_in[5];   // [512]
    const float* Wo  = (const float*)d_in[6];   // [512,1]
    const float* bo  = (const float*)d_in[7];   // [1]
    float* out = (float*)d_out;                 // [16,1024]

    char* ws = (char*)d_ws;
    unsigned short* web    = (unsigned short*)(ws);                 // 1 MB
    float*          decp   = (float*)(ws + (1u << 20));             // 32 KB
    float*          scores = (float*)(ws + (1u << 20) + 32768u);    // 256 KB
    float*          attn   = (float*)(ws + (1u << 20) + 32768u + 262144u);
    float*          part   = (float*)(ws + (1u << 20) + 32768u + 2u * 262144u); // 1 MB

    webprep_kernel<<<1024, 32, 0, stream>>>(We, web);
    decproj_kernel<<<B_, 256, 0, stream>>>(dh, Wd, bd, decp);
    scores_kernel<<<(B_ * S_) / TM, 256, 0, stream>>>(enc, web, be, decp, Wo, bo, scores);
    softmax_kernel<<<B_, 256, 0, stream>>>(scores, attn);
    ctx_partial_kernel<<<dim3(E_ / 256, S_ / 256, B_), 256, 0, stream>>>(enc, attn, part);
    ctx_reduce_kernel<<<(B_ * E_) / 256, 256, 0, stream>>>(part, out);
}
